// GMA_MoE_Layer_80599356277456
// MI455X (gfx1250) — compile-verified, matrix operand order checked by reading, not run on hardware
//
#include <hip/hip_runtime.h>
#include <hip/hip_bf16.h>
#include <math.h>

#define B_ 4
#define S_ 2048
#define D_ 1024
#define E_ 8
#define K_ 2
#define H_ 2048
#define NTOK (B_ * S_)          // 8192 tokens
#define NASSIGN (NTOK * K_)     // 16384 (token, expert) assignments

typedef __bf16 bf16;
typedef __attribute__((ext_vector_type(16))) __bf16 v16bf;
typedef __attribute__((ext_vector_type(8)))  __bf16 v8bf;
typedef __attribute__((ext_vector_type(8)))  float  v8f;
typedef __attribute__((ext_vector_type(4)))  unsigned int u32x4;
typedef __attribute__((ext_vector_type(8)))  int i32x8;
typedef __attribute__((ext_vector_type(4)))  int i32x4;

static __device__ __forceinline__ v16bf cat16(v8bf lo, v8bf hi) {
  v16bf r;
#pragma unroll
  for (int i = 0; i < 8; ++i) { r[i] = lo[i]; r[i + 8] = hi[i]; }
  return r;
}

// ---------------------------------------------------------------------------
// Kernel 0: fp32 -> bf16 weight conversion (one-shot HBM read; bf16 weights
// then fit entirely in the 192MB L2 for the GEMM phase).
// ---------------------------------------------------------------------------
__global__ __launch_bounds__(256) void cvt_bf16_kernel(const float* __restrict__ src,
                                                       bf16* __restrict__ dst, int n8) {
  int i = blockIdx.x * blockDim.x + threadIdx.x;
  if (i >= n8) return;
  const float4* s = (const float4*)src;
  float4 a = s[2 * i], b = s[2 * i + 1];
  union { bf16 h[8]; uint4 u; } p;
  p.h[0] = (bf16)a.x; p.h[1] = (bf16)a.y; p.h[2] = (bf16)a.z; p.h[3] = (bf16)a.w;
  p.h[4] = (bf16)b.x; p.h[5] = (bf16)b.y; p.h[6] = (bf16)b.z; p.h[7] = (bf16)b.w;
  ((uint4*)dst)[i] = p.u;
}

__global__ void zero_counts_kernel(int* counts) {
  if (threadIdx.x < E_) counts[threadIdx.x] = 0;
}

// ---------------------------------------------------------------------------
// Kernel 1: fused RMSNorm + RoPE + router (softmax/top-2) + expert routing.
// One block (256 thr = 8 wave32) per token; 4 elements per thread.
// ---------------------------------------------------------------------------
__global__ __launch_bounds__(256) void prep_kernel(
    const float* __restrict__ x, const float* __restrict__ norm_w,
    const float* __restrict__ rW, bf16* __restrict__ hidden,
    int* __restrict__ counts, int* __restrict__ tok_list,
    int* __restrict__ expert_for, int* __restrict__ slot_for,
    float* __restrict__ gate_for) {
  const int t   = blockIdx.x;
  const int tid = threadIdx.x;
  const int s   = t & (S_ - 1);
  const float* xr = x + (size_t)t * D_;

  float x0 = xr[tid], x1 = xr[tid + 256], x2 = xr[tid + 512], x3 = xr[tid + 768];

  // RMS over D via wave32 shuffle + LDS tree
  float ss = x0 * x0 + x1 * x1 + x2 * x2 + x3 * x3;
#pragma unroll
  for (int off = 16; off > 0; off >>= 1) ss += __shfl_down(ss, off, 32);
  __shared__ float red[8];
  __shared__ float rbc;
  const int wv = tid >> 5, lane = tid & 31;
  if (lane == 0) red[wv] = ss;
  __syncthreads();
  if (tid == 0) {
    float tot = 0.f;
    for (int i = 0; i < 8; ++i) tot += red[i];
    rbc = rsqrtf(tot / (float)D_ + 1e-6f);
  }
  __syncthreads();
  const float rms = rbc;
  float n0 = x0 * rms * norm_w[tid];
  float n1 = x1 * rms * norm_w[tid + 256];
  float n2 = x2 * rms * norm_w[tid + 512];
  float n3 = x3 * rms * norm_w[tid + 768];

  // RoPE: pairs (i, i+512); this thread owns pairs i=tid and i=tid+256
  const float LOG_BASE = 9.210340371976184f; // ln(10000)
  const float fs = (float)s;
  float invA = __expf(-((float)tid * (1.0f / 512.0f)) * LOG_BASE);
  float invB = __expf(-((float)(tid + 256) * (1.0f / 512.0f)) * LOG_BASE);
  float aa = fs * invA, ab = fs * invB;
  float ca = cosf(aa), sa = sinf(aa);
  float cb = cosf(ab), sb = sinf(ab);
  float h0 = n0 * ca - n2 * sa;
  float h2 = n0 * sa + n2 * ca;
  float h1 = n1 * cb - n3 * sb;
  float h3 = n1 * sb + n3 * cb;

  bf16* hr = hidden + (size_t)t * D_;
  hr[tid] = (bf16)h0; hr[tid + 256] = (bf16)h1;
  hr[tid + 512] = (bf16)h2; hr[tid + 768] = (bf16)h3;

  // Router logits: 8 dot products over D (router_W is 32KB -> L2/L0 resident)
  float p[E_];
#pragma unroll
  for (int e = 0; e < E_; ++e) {
    const float* w = rW + e * D_;
    p[e] = h0 * w[tid] + h1 * w[tid + 256] + h2 * w[tid + 512] + h3 * w[tid + 768];
  }
#pragma unroll
  for (int e = 0; e < E_; ++e)
#pragma unroll
    for (int off = 16; off > 0; off >>= 1) p[e] += __shfl_down(p[e], off, 32);
  __shared__ float red2[8][E_];
  if (lane == 0)
    for (int e = 0; e < E_; ++e) red2[wv][e] = p[e];
  __syncthreads();

  if (tid == 0) {
    float lg[E_];
    for (int e = 0; e < E_; ++e) {
      float a = 0.f;
      for (int w = 0; w < 8; ++w) a += red2[w][e];
      lg[e] = a;
    }
    float mx = lg[0];
    for (int e = 1; e < E_; ++e) mx = fmaxf(mx, lg[e]);
    float se = 0.f, pr[E_];
    for (int e = 0; e < E_; ++e) { pr[e] = __expf(lg[e] - mx); se += pr[e]; }
    float inv = 1.0f / se;
    for (int e = 0; e < E_; ++e) pr[e] *= inv;
    int i1 = 0;
    for (int e = 1; e < E_; ++e) if (pr[e] > pr[i1]) i1 = e;
    int i2 = (i1 == 0) ? 1 : 0;
    for (int e = 0; e < E_; ++e) if (e != i1 && pr[e] > pr[i2]) i2 = e;
    float v1 = pr[i1], v2 = pr[i2];
    float den = fmaxf(v1 + v2, 1e-8f);
    int s1 = atomicAdd(&counts[i1], 1);
    int s2 = atomicAdd(&counts[i2], 1);
    tok_list[i1 * NTOK + s1] = t;
    tok_list[i2 * NTOK + s2] = t;
    expert_for[2 * t] = i1;     expert_for[2 * t + 1] = i2;
    slot_for[2 * t] = s1;       slot_for[2 * t + 1] = s2;
    gate_for[2 * t] = v1 / den; gate_for[2 * t + 1] = v2 / den;
  }
}

// ---------------------------------------------------------------------------
// Kernel 2: sparse expert FFN. Block = (expert, 16-token tile), 8 wave32.
// Hidden tile gathered into LDS by the Tensor Data Mover (gather-mode D#,
// 16 x 16-bit row indices); per 64-wide h-chunk: waves 0-3 -> G tiles,
// waves 4-7 -> U tiles (WMMA bf16 over K=D), SiLU*U into LDS bf16, then all
// waves accumulate their 128 d-columns of the output with WMMA over h.
// ---------------------------------------------------------------------------
__global__ __launch_bounds__(256) void ffn_kernel(
    const bf16* __restrict__ hidden, const bf16* __restrict__ w1b,
    const bf16* __restrict__ w3b, const bf16* __restrict__ w2b,
    const int* __restrict__ counts, const int* __restrict__ tok_list,
    float* __restrict__ out_buf) {
  const int e = blockIdx.x;
  const int tile = blockIdx.y;
  const int cnt = counts[e];
  const int row0 = tile * 16;
  if (row0 >= cnt) return;
  int offset = 0;
  for (int j = 0; j < e; ++j) offset += counts[j];

  __shared__ bf16  sH[16][D_];     // 32 KB hidden tile
  __shared__ float sG[16][64];     // 4 KB
  __shared__ float sU[16][64];     // 4 KB
  __shared__ bf16  sAct[16][64];   // 2 KB

  const int tid = threadIdx.x;

#if __has_builtin(__builtin_amdgcn_tensor_load_to_lds) && defined(__gfx1250__)
  // --- TDM gather: 16 scattered token rows (1024 x bf16 each) -> LDS ---
  if (tid < 32) {  // wave 0 issues the DMA; descriptor built from SGPRs
    int idx[16];
#pragma unroll
    for (int i = 0; i < 16; ++i) {
      int r = row0 + i;
      idx[i] = tok_list[e * NTOK + ((r < cnt) ? r : row0)];
    }
    unsigned long long ga = (unsigned long long)(uintptr_t)hidden;
    unsigned int lds_addr = (unsigned int)(uintptr_t)(&sH[0][0]);
    u32x4 g0;
    g0[0] = 1u | (0u << 30) | (1u << 31);          // count=1, 16-bit idx, gather
    g0[1] = lds_addr;                              // D#.lds_addr
    g0[2] = (unsigned int)(ga & 0xFFFFFFFFu);      // global_addr[31:0]
    g0[3] = (unsigned int)((ga >> 32) & 0x1FFFFFFu) | (2u << 30);  // addr[56:32], type=2
    i32x8 g1;
    g1[0] = (1 << 16);                             // wg_mask=0, data_size=2B
    g1[1] = (D_ & 0xFFFF) << 16;                   // no atomic barrier; tensor_dim0 lo
    g1[2] = (NTOK & 0xFFFF) << 16;                 // tensor_dim0 hi=0, tensor_dim1 lo
    g1[3] = (D_ & 0xFFFF) << 16;                   // tensor_dim1 hi=0, tile_dim0 = 1024
    g1[4] = 16;                                    // tile_dim1 = 16 gathered rows
    g1[5] = D_;                                    // tensor_dim0_stride = 1024
    g1[6] = 0;
    g1[7] = 0;
    i32x4 g2, g3;
#pragma unroll
    for (int i = 0; i < 4; ++i) {
      g2[i] = (idx[2 * i] & 0xFFFF) | (idx[2 * i + 1] << 16);
      g3[i] = (idx[8 + 2 * i] & 0xFFFF) | (idx[8 + 2 * i + 1] << 16);
    }
    i32x8 g4;
#pragma unroll
    for (int i = 0; i < 8; ++i) g4[i] = 0;         // unused trailing descriptor
    __builtin_amdgcn_tensor_load_to_lds(g0, g1, g2, g3, g4, 0);
    __builtin_amdgcn_s_wait_tensorcnt(0);
  }
  __syncthreads();
#else
  // --- fallback: manual gather copy ---
  __shared__ int sTok[16];
  if (tid < 16) {
    int r = row0 + tid;
    sTok[tid] = tok_list[e * NTOK + ((r < cnt) ? r : row0)];
  }
  __syncthreads();
  {
    int m = tid >> 4, cp = tid & 15;
    const uint4* src = (const uint4*)(hidden + (size_t)sTok[m] * D_);
    uint4* dst = (uint4*)(&sH[m][0]);
#pragma unroll
    for (int i = 0; i < 8; ++i) dst[cp * 8 + i] = src[cp * 8 + i];
  }
  __syncthreads();
#endif

  const int wv   = tid >> 5;
  const int lane = tid & 31;
  const int col  = lane & 15;
  const int hf   = lane >> 4;

  v8f acc[8];
#pragma unroll
  for (int i = 0; i < 8; ++i)
#pragma unroll
    for (int j = 0; j < 8; ++j) acc[i][j] = 0.0f;

  const bf16* Wgu = (wv < 4) ? w1b : w3b;
  const int ht = wv & 3;
  const size_t eH = (size_t)e * H_;

  for (int hc0 = 0; hc0 < H_; hc0 += 64) {
    // ---- matmul 1: one 16x16 G or U tile per wave, K = D = 1024
    const int hcol = hc0 + ht * 16 + col;
    const bf16* wrow = Wgu + (eH + (size_t)hcol) * (size_t)D_;
    if (hc0 + 64 < H_)  // prefetch next chunk's weight row (global_prefetch_b8)
      __builtin_prefetch((const void*)(wrow + 64 * (size_t)D_), 0, 0);
    v8f c;
#pragma unroll
    for (int j = 0; j < 8; ++j) c[j] = 0.0f;
#pragma unroll 4
    for (int k0 = 0; k0 < D_; k0 += 32) {
      v8bf alo = *(const v8bf*)(&sH[col][k0 + 8 * hf]);
      v8bf ahi = *(const v8bf*)(&sH[col][k0 + 16 + 8 * hf]);
      v8bf blo = *(const v8bf*)(wrow + k0 + 8 * hf);
      v8bf bhi = *(const v8bf*)(wrow + k0 + 16 + 8 * hf);
      c = __builtin_amdgcn_wmma_f32_16x16x32_bf16(false, cat16(alo, ahi),
                                                  false, cat16(blo, bhi),
                                                  (short)0, c, false, false);
    }
    __syncthreads();  // previous iteration fully consumed sG/sU/sAct
    {
      float* dst = (wv < 4) ? &sG[0][0] : &sU[0][0];
#pragma unroll
      for (int j = 0; j < 8; ++j)
        dst[(j + 8 * hf) * 64 + ht * 16 + col] = c[j];
    }
    __syncthreads();
    // ---- SiLU(g) * u -> bf16 act tile (fast v_rcp_f32, not IEEE division)
    {
      const int base = tid * 4;
#pragma unroll
      for (int i = 0; i < 4; ++i) {
        int li = base + i;
        int m = li >> 6, hl = li & 63;
        float g = sG[m][hl];
        float u = sU[m][hl];
        float sig = __builtin_amdgcn_rcpf(1.0f + __expf(-g));
        sAct[m][hl] = (bf16)(g * sig * u);
      }
    }
    __syncthreads();
    // ---- matmul 2: out[16 x 128(this wave)] += act(16x64) * w2[:,h-chunk]^T
#pragma unroll
    for (int kk = 0; kk < 64; kk += 32) {
      v8bf alo = *(const v8bf*)(&sAct[col][kk + 8 * hf]);
      v8bf ahi = *(const v8bf*)(&sAct[col][kk + 16 + 8 * hf]);
      v16bf a = cat16(alo, ahi);
#pragma unroll
      for (int dt = 0; dt < 8; ++dt) {
        const int d = wv * 128 + dt * 16 + col;
        const bf16* w2row = w2b + ((size_t)e * D_ + d) * (size_t)H_ + hc0;
        v8bf blo = *(const v8bf*)(w2row + kk + 8 * hf);
        v8bf bhi = *(const v8bf*)(w2row + kk + 16 + 8 * hf);
        acc[dt] = __builtin_amdgcn_wmma_f32_16x16x32_bf16(false, a,
                                                          false, cat16(blo, bhi),
                                                          (short)0, acc[dt],
                                                          false, false);
      }
    }
  }

  // Epilogue: write per-assignment output rows (fast path for full tiles).
  float* obase = out_buf + (size_t)(offset + row0) * D_ + wv * 128 + col;
  if (row0 + 16 <= cnt) {
#pragma unroll
    for (int dt = 0; dt < 8; ++dt)
#pragma unroll
      for (int j = 0; j < 8; ++j)
        obase[(size_t)(j + 8 * hf) * D_ + dt * 16] = acc[dt][j];
  } else {
#pragma unroll
    for (int dt = 0; dt < 8; ++dt)
#pragma unroll
      for (int j = 0; j < 8; ++j) {
        int m = j + 8 * hf;
        if (row0 + m < cnt)
          obase[(size_t)m * D_ + dt * 16] = acc[dt][j];
      }
  }
}

// ---------------------------------------------------------------------------
// Kernel 3: per-token gated combine + residual (deterministic: each token
// sums exactly its own two expert rows).
// ---------------------------------------------------------------------------
__global__ __launch_bounds__(256) void gather_kernel(
    const float* __restrict__ x, const float* __restrict__ out_buf,
    const int* __restrict__ counts, const int* __restrict__ expert_for,
    const int* __restrict__ slot_for, const float* __restrict__ gate_for,
    float* __restrict__ out) {
  const int t = blockIdx.x;
  __shared__ int   rr[2];
  __shared__ float gg[2];
  if (threadIdx.x < 2) {
    int k = threadIdx.x;
    int e = expert_for[2 * t + k];
    int off = 0;
    for (int j = 0; j < e; ++j) off += counts[j];
    rr[k] = off + slot_for[2 * t + k];
    gg[k] = gate_for[2 * t + k];
  }
  __syncthreads();
  const float g0 = gg[0], g1 = gg[1];
  const float* r0 = out_buf + (size_t)rr[0] * D_;
  const float* r1 = out_buf + (size_t)rr[1] * D_;
  const float* xr = x + (size_t)t * D_;
  float* o = out + (size_t)t * D_;
  const int d = threadIdx.x * 4;
  float4 a  = *(const float4*)(xr + d);
  float4 b0 = *(const float4*)(r0 + d);
  float4 b1 = *(const float4*)(r1 + d);
  float4 r;
  r.x = a.x + g0 * b0.x + g1 * b1.x;
  r.y = a.y + g0 * b0.y + g1 * b1.y;
  r.z = a.z + g0 * b0.z + g1 * b1.z;
  r.w = a.w + g0 * b0.w + g1 * b1.w;
  *(float4*)(o + d) = r;
}

// ---------------------------------------------------------------------------
extern "C" void kernel_launch(void* const* d_in, const int* in_sizes, int n_in,
                              void* d_out, int out_size, void* d_ws, size_t ws_size,
                              hipStream_t stream) {
  const float* x      = (const float*)d_in[0];
  const float* norm_w = (const float*)d_in[1];
  const float* rW     = (const float*)d_in[2];
  const float* w1     = (const float*)d_in[3];
  const float* w3     = (const float*)d_in[4];
  const float* w2     = (const float*)d_in[5];
  float* out = (float*)d_out;
  (void)in_sizes; (void)n_in; (void)out_size; (void)ws_size;

  char* ws = (char*)d_ws;
  size_t off = 0;
  auto alloc = [&](size_t bytes) -> char* {
    char* p = ws + off;
    off += (bytes + 255) & ~(size_t)255;
    return p;
  };
  bf16*  hidden     = (bf16*)alloc((size_t)NTOK * D_ * sizeof(bf16));
  bf16*  w1b        = (bf16*)alloc((size_t)E_ * H_ * D_ * sizeof(bf16));
  bf16*  w3b        = (bf16*)alloc((size_t)E_ * H_ * D_ * sizeof(bf16));
  bf16*  w2b        = (bf16*)alloc((size_t)E_ * D_ * H_ * sizeof(bf16));
  float* out_buf    = (float*)alloc((size_t)NASSIGN * D_ * sizeof(float));
  int*   counts     = (int*)alloc(E_ * sizeof(int));
  int*   tok_list   = (int*)alloc((size_t)E_ * NTOK * sizeof(int));
  int*   expert_for = (int*)alloc((size_t)NASSIGN * sizeof(int));
  int*   slot_for   = (int*)alloc((size_t)NASSIGN * sizeof(int));
  float* gate_for   = (float*)alloc((size_t)NASSIGN * sizeof(float));

  const int n8 = (E_ * H_ * D_) / 8;
  const int cgrid = (n8 + 255) / 256;
  cvt_bf16_kernel<<<cgrid, 256, 0, stream>>>(w1, w1b, n8);
  cvt_bf16_kernel<<<cgrid, 256, 0, stream>>>(w3, w3b, n8);
  cvt_bf16_kernel<<<cgrid, 256, 0, stream>>>(w2, w2b, n8);
  zero_counts_kernel<<<1, 32, 0, stream>>>(counts);
  prep_kernel<<<NTOK, 256, 0, stream>>>(x, norm_w, rW, hidden, counts, tok_list,
                                        expert_for, slot_for, gate_for);
  ffn_kernel<<<dim3(E_, NTOK / 16), 256, 0, stream>>>(hidden, w1b, w3b, w2b,
                                                      counts, tok_list, out_buf);
  gather_kernel<<<NTOK, 256, 0, stream>>>(x, out_buf, counts, expert_for,
                                          slot_for, gate_for, out);
}
